// RoutingAssignmentV2_70265664963133
// MI455X (gfx1250) — matrix operand order repaired and checked by reading, not compiled, not run on hardware
//
#include <hip/hip_runtime.h>
#include <stdint.h>

// ---------------------------------------------------------------------------
// Problem constants (from reference): B=32, N=2048, D=512, H=512, C=512
// ---------------------------------------------------------------------------
constexpr int kB = 32;
constexpr int kN = 2048;
constexpr int kD = 512;
constexpr int kH = 512;
constexpr int kC = 512;   // max_cluster = N * 0.25

typedef float  v8f    __attribute__((ext_vector_type(8)));
typedef __bf16 v16bf  __attribute__((ext_vector_type(16)));
typedef __bf16 bf16x8 __attribute__((ext_vector_type(8)));

template<class A, class B> struct is_same_t { static constexpr bool v = false; };
template<class A>          struct is_same_t<A, A> { static constexpr bool v = true; };

// ---------------------------------------------------------------------------
// Wave32 reductions
// ---------------------------------------------------------------------------
__device__ __forceinline__ float wave_max(float v) {
  #pragma unroll
  for (int o = 16; o > 0; o >>= 1) v = fmaxf(v, __shfl_xor(v, o, 32));
  return v;
}
__device__ __forceinline__ float wave_sum(float v) {
  #pragma unroll
  for (int o = 16; o > 0; o >>= 1) v += __shfl_xor(v, o, 32);
  return v;
}

// ---------------------------------------------------------------------------
// LDS staging helpers
// ---------------------------------------------------------------------------
// Synchronous: 16 contiguous K-elements, f32 -> bf16 convert through VALU.
__device__ __forceinline__ void stage16(__bf16* dst, const float* src) {
  const float4* s4 = (const float4*)src;
  float4 f0 = s4[0], f1 = s4[1], f2 = s4[2], f3 = s4[3];
  bf16x8 h0, h1;
  h0[0] = (__bf16)f0.x; h0[1] = (__bf16)f0.y; h0[2] = (__bf16)f0.z; h0[3] = (__bf16)f0.w;
  h0[4] = (__bf16)f1.x; h0[5] = (__bf16)f1.y; h0[6] = (__bf16)f1.z; h0[7] = (__bf16)f1.w;
  h1[0] = (__bf16)f2.x; h1[1] = (__bf16)f2.y; h1[2] = (__bf16)f2.z; h1[3] = (__bf16)f2.w;
  h1[4] = (__bf16)f3.x; h1[5] = (__bf16)f3.y; h1[6] = (__bf16)f3.z; h1[7] = (__bf16)f3.w;
  *(bf16x8*)dst       = h0;
  *(bf16x8*)(dst + 8) = h1;
}
// Async (ASYNCcnt-tracked) 16B global->LDS DMA; per-lane LDS dest = low 32
// bits of the generic pointer (flat LDS aperture).
__device__ __forceinline__ void async_ld16B(__bf16* dst, const __bf16* src) {
  uint32_t l = (uint32_t)(uintptr_t)dst;
  uint64_t g = (uint64_t)(uintptr_t)src;
  asm volatile("global_load_async_to_lds_b128 %0, %1, off" :: "v"(l), "v"(g) : "memory");
}
__device__ __forceinline__ void async_ld32B(__bf16* dst, const __bf16* src) {
  uint32_t l = (uint32_t)(uintptr_t)dst;
  uint64_t g = (uint64_t)(uintptr_t)src;
  asm volatile("global_load_async_to_lds_b128 %0, %1, off"           :: "v"(l), "v"(g) : "memory");
  asm volatile("global_load_async_to_lds_b128 %0, %1, off offset:16" :: "v"(l), "v"(g) : "memory");
}
template<int N>
__device__ __forceinline__ void wait_async() {
  if constexpr (N == 0)      asm volatile("s_wait_asynccnt 0x0" ::: "memory");
  else if constexpr (N == 4) asm volatile("s_wait_asynccnt 0x4" ::: "memory");
  else                       asm volatile("s_wait_asynccnt 0x6" ::: "memory");
}

// ---------------------------------------------------------------------------
// Batched WMMA GEMM:  C[M x Ncols] = A[M x K] * B[Ncols x K]^T  (bf16 MACs,
// f32 accum).  A row-major (M x K); B stored Ncols x K (K-contiguous per
// output column).  TRANS_A: A element (m,k) read from A[k*lda + m].
// Tile: 128(M) x 256(N) x 32(K); 256 threads = 8 waves in 2(m) x 4(n); each
// wave owns a 64x64 subtile = 4x4 fragments -> 16 v_wmma_f32_16x16x32_bf16
// per K-step vs 16 ds_load_b128 fragment loads (1:1).
// Double-buffered LDS; bf16 operands staged with global_load_async_to_lds
// (ASYNCcnt pipeline), f32 operands converted through VALU during staging.
// Epilogue: optional bias+relu (per output column), accumulate into f32 C,
// f32 / bf16 / transposed-bf16 (b128-vectorized) writes.
// All problem dims are multiples of 128/256; K multiple of 32: no edges.
// ---------------------------------------------------------------------------
constexpr int BM = 128, BN = 256, BK = 32, LDSP = BK + 8;  // +8 halfs pad

template<bool TRANS_A, typename TA, typename TB,
         bool BIAS_RELU, bool ACCUM, bool OUT_F32, bool OUT_BF, bool OUT_T>
__global__ __launch_bounds__(256)
void gemm_wmma_bf16(const TA* __restrict__ Ag, long long strideA, int lda,
                    const TB* __restrict__ Bg, long long strideB, int ldb,
                    const float* __restrict__ bias,
                    float*  __restrict__ Cf, long long strideCf,
                    __bf16* __restrict__ Cb, long long strideCb,
                    __bf16* __restrict__ Ct, long long strideCt, int ldt,
                    int ldc, int K)
{
  constexpr bool ASYNC_A = is_same_t<TA, __bf16>::v && !TRANS_A;
  constexpr bool ASYNC_B = is_same_t<TB, __bf16>::v;
  constexpr int  NPEND   = (ASYNC_A ? 2 : 0) + (ASYNC_B ? 4 : 0);

  __shared__ __bf16 As[2][BM][LDSP];
  __shared__ __bf16 Bs[2][BN][LDSP];

  const int tid  = threadIdx.x;
  const int lane = tid & 31;
  const int wave = tid >> 5;
  const int wm = wave >> 2;          // 0..1  (64 rows each)
  const int wn = wave & 3;           // 0..3  (64 cols each)
  const int lo = lane & 15;
  const int hi = lane >> 4;

  const int tileN = blockIdx.x * BN;
  const int tileM = blockIdx.y * BM;
  const size_t batch = blockIdx.z;

  const TA* Ap = Ag + batch * (size_t)strideA;
  const TB* Bp = Bg + batch * (size_t)strideB;

  v8f acc[4][4];
  #pragma unroll
  for (int i = 0; i < 4; ++i)
    #pragma unroll
    for (int j = 0; j < 4; ++j)
      #pragma unroll
      for (int e = 0; e < 8; ++e) acc[i][j][e] = 0.0f;

  // ---- staging (per thread) ----
  auto stageA = [&](int buf, int k0) {
    if constexpr (TRANS_A) {
      const int kk = tid >> 3;             // 0..31
      const int ms = (tid & 7) * 16;       // 0..112
      const TA* src = Ap + (size_t)(k0 + kk) * lda + tileM + ms;
      #pragma unroll
      for (int i = 0; i < 16; ++i) As[buf][ms + i][kk] = (__bf16)(float)src[i];
    } else {
      const int row = tid >> 1;            // 0..127
      const int kp  = (tid & 1) * 16;      // 0 or 16
      if constexpr (ASYNC_A) {
        async_ld32B(&As[buf][row][kp],
                    (const __bf16*)Ap + (size_t)(tileM + row) * lda + k0 + kp);
      } else {
        const float* src = (const float*)Ap + (size_t)(tileM + row) * lda + k0 + kp;
        stage16(&As[buf][row][kp], src);
        if (k0 + BK < K) __builtin_prefetch(src + BK, 0, 1);  // global_prefetch_b8
      }
    }
  };
  auto stageB = [&](int buf, int k0) {
    if constexpr (ASYNC_B) {
      const int r = tid >> 2;              // 0..63
      const int c = (tid & 3) * 8;         // 0,8,16,24 halfs (16B chunks)
      #pragma unroll
      for (int p = 0; p < 4; ++p)          // 4 x 64 rows
        async_ld16B(&Bs[buf][p * 64 + r][c],
                    (const __bf16*)Bp + (size_t)(tileN + p * 64 + r) * ldb + k0 + c);
    } else {
      const int row = tid >> 1;            // 0..127
      const int kp  = (tid & 1) * 16;
      #pragma unroll
      for (int p = 0; p < 2; ++p) {
        const float* src = (const float*)Bp + (size_t)(tileN + p * 128 + row) * ldb + k0 + kp;
        stage16(&Bs[buf][p * 128 + row][kp], src);
        if (k0 + BK < K) __builtin_prefetch(src + BK, 0, 1);
      }
    }
  };

  const int nk = K / BK;
  stageA(0, 0);
  stageB(0, 0);

  for (int i = 0; i < nk; ++i) {
    const int  cur  = i & 1;
    const bool more = (i + 1) < nk;
    if (more) {                         // prefetch next tile into other buffer
      stageA(cur ^ 1, (i + 1) * BK);
      stageB(cur ^ 1, (i + 1) * BK);
    }
    if constexpr (NPEND > 0) {          // current tile's async DMAs complete
      if (more) wait_async<NPEND>(); else wait_async<0>();
    }
    __syncthreads();                    // all waves' staging of `cur` visible

    // ---- fragments (ISA 16-bit A / B wave32 layouts) ----
    v16bf af[4], bfv[4];
    #pragma unroll
    for (int fm = 0; fm < 4; ++fm) {
      const __bf16* p = &As[cur][wm * 64 + fm * 16 + lo][hi * 8];
      bf16x8 l0 = *(const bf16x8*)p;          // K = 8*hi .. +7
      bf16x8 l1 = *(const bf16x8*)(p + 16);   // K = 16+8*hi .. +7
      af[fm] = __builtin_shufflevector(l0, l1, 0,1,2,3,4,5,6,7,8,9,10,11,12,13,14,15);
    }
    #pragma unroll
    for (int fn = 0; fn < 4; ++fn) {
      const __bf16* p = &Bs[cur][wn * 64 + fn * 16 + lo][hi * 8];
      bf16x8 l0 = *(const bf16x8*)p;          // K = 8*hi .. +7
      bf16x8 l1 = *(const bf16x8*)(p + 16);   // K = 16+8*hi .. +7
      bfv[fn] = __builtin_shufflevector(l0, l1, 0,1,2,3,4,5,6,7,8,9,10,11,12,13,14,15);
    }
    #pragma unroll
    for (int fm = 0; fm < 4; ++fm)
      #pragma unroll
      for (int fn = 0; fn < 4; ++fn)
        acc[fm][fn] = __builtin_amdgcn_wmma_f32_16x16x32_bf16(
            false, af[fm], false, bfv[fn], (short)0, acc[fm][fn], false, false);
    __syncthreads();                    // `cur` free to restage at i+2
  }

  // ---- epilogue: D layout -> M_local = r + 8*hi, N_local = lo ----
  #pragma unroll
  for (int fm = 0; fm < 4; ++fm) {
    #pragma unroll
    for (int fn = 0; fn < 4; ++fn) {
      const int m0 = tileM + wm * 64 + fm * 16 + hi * 8;
      const int n  = tileN + wn * 64 + fn * 16 + lo;
      float vv[8];
      #pragma unroll
      for (int r = 0; r < 8; ++r) {
        float v = acc[fm][fn][r];
        if (BIAS_RELU) { v += bias[n]; v = fmaxf(v, 0.0f); }
        if (ACCUM) v += Cf[batch * (size_t)strideCf + (size_t)(m0 + r) * ldc + n];
        vv[r] = v;
      }
      if (OUT_F32) {
        #pragma unroll
        for (int r = 0; r < 8; ++r)
          Cf[batch * (size_t)strideCf + (size_t)(m0 + r) * ldc + n] = vv[r];
      }
      if (OUT_BF) {
        #pragma unroll
        for (int r = 0; r < 8; ++r)
          Cb[batch * (size_t)strideCb + (size_t)(m0 + r) * ldc + n] = (__bf16)vv[r];
      }
      if (OUT_T) {            // 8 consecutive m per lane -> one b128 store
        bf16x8 t8;
        #pragma unroll
        for (int r = 0; r < 8; ++r) t8[r] = (__bf16)vv[r];
        *(bf16x8*)(Ct + batch * (size_t)strideCt + (size_t)n * ldt + m0) = t8;
      }
    }
  }
}

// ---------------------------------------------------------------------------
// cluster_num[b] = floor(0.25 * sum(mask[b,:])) == count >> 2
// ---------------------------------------------------------------------------
__global__ __launch_bounds__(256)
void count_kernel(const unsigned char* __restrict__ mask, int* __restrict__ ncl)
{
  __shared__ int red[8];
  const int b = blockIdx.x;
  const int t = threadIdx.x;
  int cnt = 0;
  for (int i = t; i < kN; i += 256) cnt += mask[(size_t)b * kN + i] ? 1 : 0;
  #pragma unroll
  for (int o = 16; o > 0; o >>= 1) cnt += __shfl_xor(cnt, o, 32);
  if ((t & 31) == 0) red[t >> 5] = cnt;
  __syncthreads();
  if (t == 0) {
    int s = 0;
    #pragma unroll
    for (int i = 0; i < 8; ++i) s += red[i];
    ncl[b] = s >> 2;
  }
}

// ---------------------------------------------------------------------------
// threefry2x32 (key = jax.random.key(42) -> (0, 42)) + erfinv -> N(0,1)
// ---------------------------------------------------------------------------
__device__ __forceinline__ uint32_t rotl32(uint32_t x, int r) {
  return (x << r) | (x >> (32 - r));
}
__device__ __forceinline__ void threefry2x32(uint32_t k0, uint32_t k1,
                                             uint32_t x0, uint32_t x1,
                                             uint32_t& o0, uint32_t& o1)
{
  const int R[8] = {13, 15, 26, 6, 17, 29, 16, 24};
  uint32_t ks[3] = {k0, k1, 0x1BD11BDAu ^ k0 ^ k1};
  x0 += ks[0]; x1 += ks[1];
  #pragma unroll
  for (int i = 0; i < 5; ++i) {
    #pragma unroll
    for (int j = 0; j < 4; ++j) {
      x0 += x1;
      x1 = rotl32(x1, R[(i & 1) * 4 + j]);
      x1 ^= x0;
    }
    x0 += ks[(i + 1) % 3];
    x1 += ks[(i + 2) % 3] + (uint32_t)(i + 1);
  }
  o0 = x0; o1 = x1;
}
__device__ __forceinline__ float bits_to_normal(uint32_t bits)
{
  float u = (float)(bits >> 9) * (1.0f / 8388608.0f);       // [0,1)
  const float lov = -0.99999994f;                            // nextafter(-1,0)
  float x = u * (1.0f - lov) + lov;                          // (-1,1)
  float w = -logf(fmaxf((1.0f - x) * (1.0f + x), 1e-37f));
  float p;
  if (w < 5.0f) {
    w -= 2.5f;
    p = 2.81022636e-08f;
    p = 3.43273939e-07f + p * w;  p = -3.5233877e-06f  + p * w;
    p = -4.39150654e-06f + p * w; p = 0.00021858087f   + p * w;
    p = -0.00125372503f + p * w;  p = -0.00417768164f  + p * w;
    p = 0.246640727f    + p * w;  p = 1.50140941f      + p * w;
  } else {
    w = sqrtf(w) - 3.0f;
    p = -0.000200214257f;
    p = 0.000100950558f + p * w;  p = 0.00134934322f + p * w;
    p = -0.00367342844f + p * w;  p = 0.00573950773f + p * w;
    p = -0.0076224613f  + p * w;  p = 0.00943887047f + p * w;
    p = 1.00167406f     + p * w;  p = 2.83297682f    + p * w;
  }
  return 1.41421356f * p * x;
}
// jax threefry_random_bits pairs counters (j, j+half); out0->j, out1->j+half
__global__ __launch_bounds__(256)
void init_s_kernel(float* __restrict__ s)
{
  const long long half = (long long)kB * kN * kC / 2;
  const long long i = (long long)blockIdx.x * 256 + threadIdx.x;
  if (i >= half) return;
  uint32_t o0, o1;
  threefry2x32(0u, 42u, (uint32_t)i, (uint32_t)(i + half), o0, o1);
  s[i]        = bits_to_normal(o0);
  s[i + half] = bits_to_normal(o1);
}

// ---------------------------------------------------------------------------
// Masked softmax over C=512 clusters; one wave per (b,n) row.
// Non-final: overwrite s with p (f32) and emit pT[b,c,n] (bf16, A operand of
// the xnext GEMM).  Final: write p to d_out only.
// ---------------------------------------------------------------------------
template<bool FINAL>
__global__ __launch_bounds__(256)
void softmax_kernel(float* __restrict__ s, __bf16* __restrict__ pT,
                    float* __restrict__ out, const int* __restrict__ ncl)
{
  const int lane = threadIdx.x & 31;
  const int wave = threadIdx.x >> 5;
  const long long row = (long long)blockIdx.x * 8 + wave;   // [0, kB*kN)
  const int b = (int)(row >> 11);                            // kN == 2048
  const int n = (int)(row & (kN - 1));
  const int nc = ncl[b];
  float* srow = s + row * kC;

  float v[16];
  float mx = -3.4e38f;
  #pragma unroll
  for (int i = 0; i < 16; ++i) {
    const int c = i * 32 + lane;
    float t = srow[c];
    t = (c < nc) ? t : -1.0e9f;
    v[i] = t;
    mx = fmaxf(mx, t);
  }
  mx = wave_max(mx);
  float sum = 0.0f;
  #pragma unroll
  for (int i = 0; i < 16; ++i) { v[i] = expf(v[i] - mx); sum += v[i]; }
  sum = wave_sum(sum);
  const float inv = 1.0f / sum;
  #pragma unroll
  for (int i = 0; i < 16; ++i) {
    const int c = i * 32 + lane;
    const float p = (c < nc) ? v[i] * inv : 0.0f;
    if (FINAL) {
      out[row * kC + c] = p;
    } else {
      srow[c] = p;
      pT[((size_t)b * kC + c) * kN + n] = (__bf16)p;
    }
  }
}

// ---------------------------------------------------------------------------
// squash over H per (b,c) row: v * mag2/((1+mag2)*sqrt(mag2)); emit bf16
// (this is the Bt operand of the update GEMM).
// ---------------------------------------------------------------------------
__global__ __launch_bounds__(256)
void squash_kernel(const float* __restrict__ xf, __bf16* __restrict__ xb)
{
  const int lane = threadIdx.x & 31;
  const int wave = threadIdx.x >> 5;
  const long long row = (long long)blockIdx.x * 8 + wave;   // [0, kB*kC)
  const float* src = xf + row * kH;
  float v[16];
  float ss = 0.0f;
  #pragma unroll
  for (int i = 0; i < 16; ++i) { v[i] = src[i * 32 + lane]; ss += v[i] * v[i]; }
  ss = wave_sum(ss);
  const float scale = ss / ((1.0f + ss) * sqrtf(ss));
  #pragma unroll
  for (int i = 0; i < 16; ++i)
    xb[row * kH + i * 32 + lane] = (__bf16)(v[i] * scale);
}

// ---------------------------------------------------------------------------
// Host: orchestrate the pipeline on `stream` (graph-capture safe).
// ---------------------------------------------------------------------------
extern "C" void kernel_launch(void* const* d_in, const int* in_sizes, int n_in,
                              void* d_out, int out_size, void* d_ws, size_t ws_size,
                              hipStream_t stream)
{
  const float*         x    = (const float*)d_in[0];          // (B,N,D)
  const float*         adj  = (const float*)d_in[1];          // (B,N,N)
  const float*         W    = (const float*)d_in[2];          // (D,H)
  const float*         bias = (const float*)d_in[3];          // (H,)
  const unsigned char* mask = (const unsigned char*)d_in[4];  // (B,N) bool
  float* out = (float*)d_out;                                  // (B,N,C)

  // workspace carve-up (256B aligned segments)
  char* wp = (char*)d_ws;
  auto take = [&](size_t bytes) {
    void* p = wp;
    wp += (bytes + 255) & ~(size_t)255;
    return p;
  };
  float*  s    = (float*) take((size_t)kB * kN * kC * 4);  // routing logits (f32)
  __bf16* pT   = (__bf16*)take((size_t)kB * kC * kN * 2);  // softmax(s)^T
  __bf16* tT   = (__bf16*)take((size_t)kB * kH * kN * 2);  // (xW)^T
  __bf16* z_bf = (__bf16*)take((size_t)kB * kN * kH * 2);  // z row-major
  __bf16* zT   = (__bf16*)take((size_t)kB * kH * kN * 2);  // z^T
  float*  xnf  = (float*) take((size_t)kB * kC * kH * 4);  // xnext f32
  __bf16* xnb  = (__bf16*)take((size_t)kB * kC * kH * 2);  // squash(xnext) bf16
  int*    ncl  = (int*)   take((size_t)kB * sizeof(int));

  count_kernel<<<kB, 256, 0, stream>>>(mask, ncl);
  {
    const long long half = (long long)kB * kN * kC / 2;
    init_s_kernel<<<(unsigned)((half + 255) / 256), 256, 0, stream>>>(s);
  }

  // tT[b] (H x N) = W^T (H x D) * x[b]^T : A = W (TRANS), Bt = x[b] (N x D)
  {
    dim3 g(kN / BN, kH / BM, kB);
    gemm_wmma_bf16<true, float, float, false, false, false, true, false>
      <<<g, 256, 0, stream>>>(W, 0, kH,
                              x, (long long)kN * kD, kD,
                              nullptr,
                              nullptr, 0,
                              tT, (long long)kH * kN,
                              nullptr, 0, 0,
                              /*ldc=*/kN, /*K=*/kD);
  }
  // z[b] (N x H) = relu(adj[b] (N x N) * t + bias) : A = adj, Bt = tT (H x N)
  // write both z (bf16) and z^T (bf16)
  {
    dim3 g(kH / BN, kN / BM, kB);
    gemm_wmma_bf16<false, float, __bf16, true, false, false, true, true>
      <<<g, 256, 0, stream>>>(adj, (long long)kN * kN, kN,
                              tT,  (long long)kH * kN, kN,
                              bias,
                              nullptr, 0,
                              z_bf, (long long)kN * kH,
                              zT,   (long long)kH * kN, /*ldt=*/kN,
                              /*ldc=*/kH, /*K=*/kN);
  }

  for (int it = 0; it < 3; ++it) {
    softmax_kernel<false><<<kB * kN / 8, 256, 0, stream>>>(s, pT, nullptr, ncl);
    // xnext[b] (C x H) = p^T (C x N) * z : A = pT, Bt = zT (H x N)
    {
      dim3 g(kH / BN, kC / BM, kB);
      gemm_wmma_bf16<false, __bf16, __bf16, false, false, true, false, false>
        <<<g, 256, 0, stream>>>(pT, (long long)kC * kN, kN,
                                zT, (long long)kH * kN, kN,
                                nullptr,
                                xnf, (long long)kC * kH,
                                nullptr, 0,
                                nullptr, 0, 0,
                                /*ldc=*/kH, /*K=*/kN);
    }
    squash_kernel<<<kB * kC / 8, 256, 0, stream>>>(xnf, xnb);
    // s[b] (N x C) += z (N x H) * xnext^T : A = z_bf, Bt = xnb (C x H)
    {
      dim3 g(kC / BN, kN / BM, kB);
      gemm_wmma_bf16<false, __bf16, __bf16, false, true, true, false, false>
        <<<g, 256, 0, stream>>>(z_bf, (long long)kN * kH, kH,
                                xnb,  (long long)kC * kH, kH,
                                nullptr,
                                s, (long long)kN * kC,
                                nullptr, 0,
                                nullptr, 0, 0,
                                /*ldc=*/kC, /*K=*/kH);
    }
  }

  softmax_kernel<true><<<kB * kN / 8, 256, 0, stream>>>(s, nullptr, out, ncl);
}
